// TTTLinearAttention_89258010346164
// MI455X (gfx1250) — matrix operand order repaired and checked by reading, not compiled, not run on hardware
//
#include <hip/hip_runtime.h>
#include <hip/hip_bf16.h>

#define BB 2
#define TT 4096
#define CC 2048
#define HH 16
#define DD 128
#define CHUNK 16
#define NCHUNK 256
#define ETA_BASE 0.005f
#define EPS_LN 1e-6f
#define SCALE_Q 0.08838834764831845f /* D^-0.5 */

typedef __attribute__((ext_vector_type(16))) _Float16 v16h;
typedef __attribute__((ext_vector_type(8)))  _Float16 v8h;
typedef __attribute__((ext_vector_type(8)))  float    v8f;

static __device__ __forceinline__ v8f wmma_f16(v16h a, v16h b, v8f c) {
  return __builtin_amdgcn_wmma_f32_16x16x32_f16(false, a, false, b, (short)0, c, false, false);
}

static __device__ __forceinline__ v16h mkv16(v8h lo, v8h hi) {
  return __builtin_shufflevector(lo, hi, 0,1,2,3,4,5,6,7,8,9,10,11,12,13,14,15);
}

__device__ __forceinline__ float wred(float v) {
  #pragma unroll
  for (int off = 16; off > 0; off >>= 1) v += __shfl_xor(v, off, 32);
  return v;
}

// ---------------- f32 -> f16 convert ----------------
__global__ void cvt_f32_f16(const float* __restrict__ s, _Float16* __restrict__ d, size_t n) {
  size_t i = (size_t)blockIdx.x * blockDim.x + threadIdx.x;
  size_t st = (size_t)gridDim.x * blockDim.x;
  for (; i < n; i += st) d[i] = (_Float16)s[i];
}

// ---------------- eta = ETA_BASE * sigmoid(x @ Wgate^T) ----------------
__global__ void __launch_bounds__(256)
eta_kernel(const float* __restrict__ x, const float* __restrict__ Wg,
           float* __restrict__ eta) {
  __shared__ float xs[CC];
  int row = blockIdx.x;
  int tid = threadIdx.x, lane = tid & 31, w = tid >> 5;
  for (int i = tid; i < CC; i += 256) xs[i] = x[(size_t)row * CC + i];
  __syncthreads();
  #pragma unroll
  for (int r = 0; r < 2; ++r) {
    int h = w + 8 * r;
    const float* wr = Wg + (size_t)h * CC;
    float s = 0.f;
    for (int c = lane; c < CC; c += 32) s += xs[c] * wr[c];
    s = wred(s);
    if (lane == 0) eta[(size_t)row * HH + h] = ETA_BASE / (1.f + expf(-s));
  }
}

// ---------------- WMMA GEMM: Out[M,N] = A[M,K] @ W[N,K]^T + bias ----------------
// 256 threads (8 wave32), 128x128 tile, staged K-step 64, LDS double-buffered.
// 16 WMMAs per barrier; B fragments preloaded in rolling groups of 4.
__global__ void __launch_bounds__(256)
wmma_gemm(const _Float16* __restrict__ A, const _Float16* __restrict__ W,
          const float* __restrict__ bias, float* __restrict__ Out,
          int M, int N, int K) {
  __shared__ _Float16 As[2][128 * 64];
  __shared__ _Float16 Bs[2][128 * 64];

  int tileN = blockIdx.x * 128;
  int tileM = blockIdx.y * 128;
  int tid = threadIdx.x, lane = tid & 31, w = tid >> 5;

  // cooperative staging: thread -> (row 0..127, khalf 0/32), 64B each from A and W
  int ldRow = tid >> 1;
  int ldK   = (tid & 1) << 5;
  const _Float16* gA = A + (size_t)(tileM + ldRow) * K + ldK;
  const _Float16* gB = W + (size_t)(tileN + ldRow) * K + ldK;

  int aRow = 16 * w + (lane & 15);
  int aK   = (lane >> 4) << 4;
  int bK   = (lane >> 4) << 4;
  int col16 = lane & 15;

  v8f acc[8];
  #pragma unroll
  for (int nt = 0; nt < 8; ++nt) acc[nt] = (v8f){0.f,0.f,0.f,0.f,0.f,0.f,0.f,0.f};

  // prologue: stage k-step 0
  {
    #pragma unroll
    for (int q = 0; q < 4; ++q) {
      v8h av = *(const v8h*)(gA + 8 * q);
      v8h bv = *(const v8h*)(gB + 8 * q);
      *(v8h*)(&As[0][ldRow * 64 + ldK + 8 * q]) = av;
      *(v8h*)(&Bs[0][ldRow * 64 + ldK + 8 * q]) = bv;
    }
  }
  __syncthreads();

  int nsteps = K >> 6;
  int cur = 0;
  for (int s = 0; s < nsteps; ++s) {
    v8h na[4], nb[4];
    bool more = (s + 1 < nsteps);
    if (more) {
      const _Float16* pA = gA + (size_t)(s + 1) * 64;
      const _Float16* pB = gB + (size_t)(s + 1) * 64;
      #pragma unroll
      for (int q = 0; q < 4; ++q) {
        na[q] = *(const v8h*)(pA + 8 * q);
        nb[q] = *(const v8h*)(pB + 8 * q);
      }
    }

    // two 32-wide sub-steps from buf[cur]
    #pragma unroll
    for (int ss = 0; ss < 2; ++ss) {
      const _Float16* ap = &As[cur][aRow * 64 + 32 * ss + aK];
      v16h a = mkv16(*(const v8h*)ap, *(const v8h*)(ap + 8));
      const _Float16* bp = &Bs[cur][32 * ss + bK];
      v16h bq[4];
      #pragma unroll
      for (int nt = 0; nt < 4; ++nt)
        bq[nt] = *(const v16h*)(bp + (nt * 16 + col16) * 64);
      #pragma unroll
      for (int nt = 0; nt < 4; ++nt) {
        v16h bn = *(const v16h*)(bp + ((nt + 4) * 16 + col16) * 64);
        acc[nt] = wmma_f16(a, bq[nt], acc[nt]);
        bq[nt] = bn;
      }
      #pragma unroll
      for (int nt = 0; nt < 4; ++nt)
        acc[nt + 4] = wmma_f16(a, bq[nt], acc[nt + 4]);
    }

    if (more) {
      int nxt = cur ^ 1;
      #pragma unroll
      for (int q = 0; q < 4; ++q) {
        *(v8h*)(&As[nxt][ldRow * 64 + ldK + 8 * q]) = na[q];
        *(v8h*)(&Bs[nxt][ldRow * 64 + ldK + 8 * q]) = nb[q];
      }
    }
    __syncthreads();
    cur ^= 1;
  }

  int rbase = 16 * w + ((lane >> 4) << 3) + tileM;
  #pragma unroll
  for (int nt = 0; nt < 8; ++nt) {
    int col = tileN + nt * 16 + col16;
    float bs = bias ? bias[col] : 0.f;
    #pragma unroll
    for (int j = 0; j < 8; ++j)
      Out[(size_t)(rbase + j) * N + col] = acc[nt][j] + bs;
  }
}

// ---------------- TTT scan: 1 block per (b,h), serial over 256 chunks ----------------
__global__ void __launch_bounds__(256)
ttt_scan(const float* __restrict__ qkv, const float* __restrict__ eta,
         const float* __restrict__ ln_w, const float* __restrict__ ln_b,
         const float* __restrict__ h0, const float* __restrict__ hb0,
         _Float16* __restrict__ o_h) {
  extern __shared__ char smem_raw[];
  float* Hs  = (float*)smem_raw;   // 128x128 (row k, col n)
  float* KC  = Hs  + 16384;        // 16x128
  float* VC  = KC  + 2048;
  float* QC  = VC  + 2048;
  float* Zb  = QC  + 2048;         // z, then reused for zbar
  float* G   = Zb  + 2048;
  float* CUM = G   + 2048;
  float* HB  = CUM + 2048;         // 128
  float* GW  = HB  + 128;
  float* GB  = GW  + 128;
  float* ATT = GB  + 128;          // 16x16 (ec*attn, masked)
  float* EC  = ATT + 256;          // 16
  _Float16* Hh  = (_Float16*)(EC + 16); // 128x128: Hh[n*128+k] = H[k][n] (H^T, f16)
  _Float16* KCh = Hh  + 16384;          // 16x128 f16, row-major
  _Float16* QCh = KCh + 2048;           // 16x128 f16, row-major
  _Float16* Gh  = QCh + 2048;           // 128x16 f16: Gh[n*16+i] = g[i][n] (g^T)

  int b = blockIdx.x / HH, h = blockIdx.x % HH;
  int tid = threadIdx.x, lane = tid & 31, w = tid >> 5;

  for (int i = tid; i < 16384; i += 256) {
    float v = h0[(size_t)h * 16384 + i];
    Hs[i] = v;
    int k = i >> 7, n = i & 127;
    Hh[n * DD + k] = (_Float16)v;
  }
  if (tid < 128) {
    HB[tid] = hb0[h * 128 + tid];
    GW[tid] = ln_w[h * 128 + tid];
    GB[tid] = ln_b[h * 128 + tid];
  }
  __syncthreads();

  for (int ch = 0; ch < NCHUNK; ++ch) {
    // ---- phase 1: stage kc, vc, qc (f32 + f16), ec ----
    for (int i = tid; i < 2048; i += 256) {
      int ci = i >> 7, d = i & 127;
      size_t base = (size_t)(b * TT + ch * CHUNK + ci) * (3 * CC);
      float qv = qkv[base + h * DD + d] * SCALE_Q;
      float kv = qkv[base + CC + h * DD + d];
      float vv = qkv[base + 2 * CC + h * DD + d];
      QC[i] = qv; QCh[i] = (_Float16)qv;
      KC[i] = kv; KCh[i] = (_Float16)kv;
      VC[i] = vv;
    }
    if (tid < CHUNK) EC[tid] = eta[(size_t)(b * TT + ch * CHUNK + tid) * HH + h];
    if (ch + 1 < NCHUNK && tid < CHUNK) {
      __builtin_prefetch(qkv + (size_t)(b * TT + (ch + 1) * CHUNK + tid) * (3 * CC) + h * DD, 0, 1);
    }
    __syncthreads();

    // ---- phase 2: z = kc @ H + hb  (wave w -> cols [16w,16w+16)) ----
    {
      int col = 16 * w + (lane & 15);
      int aK = (lane >> 4) << 4, bK = (lane >> 4) << 4;
      int aRow = (lane & 15);
      v16h af[4], bf[4];
      #pragma unroll
      for (int t = 0; t < 4; ++t) {
        const _Float16* ap = KCh + aRow * DD + 32 * t + aK;
        af[t] = mkv16(*(const v8h*)ap, *(const v8h*)(ap + 8));
        bf[t] = *(const v16h*)(Hh + col * DD + 32 * t + bK);
      }
      v8f acc = (v8f){0.f,0.f,0.f,0.f,0.f,0.f,0.f,0.f};
      #pragma unroll
      for (int t = 0; t < 4; ++t) acc = wmma_f16(af[t], bf[t], acc);
      int rb = (lane >> 4) << 3;
      #pragma unroll
      for (int j = 0; j < 8; ++j) Zb[(rb + j) * DD + col] = acc[j] + HB[col];
    }
    __syncthreads();

    // ---- phase 3: g = ln_l2_grad(z, vc-kc) (wave w -> rows 2w, 2w+1) ----
    #pragma unroll
    for (int rr = 0; rr < 2; ++rr) {
      int r = 2 * w + rr;
      float s1 = 0.f, s2 = 0.f;
      for (int c = lane; c < DD; c += 32) { float z = Zb[r * DD + c]; s1 += z; s2 += z * z; }
      s1 = wred(s1); s2 = wred(s2);
      float mu = s1 * (1.f / DD);
      float var = s2 * (1.f / DD) - mu * mu;
      float rstd = rsqrtf(var + EPS_LN);
      float t1 = 0.f, t2 = 0.f;
      for (int c = lane; c < DD; c += 32) {
        float z = Zb[r * DD + c];
        float xh = (z - mu) * rstd;
        float gw = GW[c];
        float dy = gw * xh + GB[c] - (VC[r * DD + c] - KC[r * DD + c]);
        float dxh = dy * gw;
        t1 += dxh; t2 += dxh * xh;
      }
      t1 = wred(t1) * (1.f / DD); t2 = wred(t2) * (1.f / DD);
      for (int c = lane; c < DD; c += 32) {
        float z = Zb[r * DD + c];
        float xh = (z - mu) * rstd;
        float gw = GW[c];
        float dy = gw * xh + GB[c] - (VC[r * DD + c] - KC[r * DD + c]);
        float dxh = dy * gw;
        float gval = rstd * (dxh - t1 - xh * t2);
        G[r * DD + c] = gval;
        Gh[c * 16 + r] = (_Float16)gval;   // g^T for WMMA B operands
      }
    }
    __syncthreads();

    // ---- phase 4: attn (wave 0) + cumsum (waves 4-7) ----
    if (w == 0) {
      int aRow = (lane & 15), aK = (lane >> 4) << 4;
      int n = (lane & 15), bK = (lane >> 4) << 4;
      v16h af[4], bf[4];
      #pragma unroll
      for (int t = 0; t < 4; ++t) {
        const _Float16* ap = QCh + aRow * DD + 32 * t + aK;
        af[t] = mkv16(*(const v8h*)ap, *(const v8h*)(ap + 8));
        bf[t] = *(const v16h*)(KCh + n * DD + 32 * t + bK); // B[k][j] = KC[j][k]
      }
      v8f acc = (v8f){0.f,0.f,0.f,0.f,0.f,0.f,0.f,0.f};
      #pragma unroll
      for (int t = 0; t < 4; ++t) acc = wmma_f16(af[t], bf[t], acc);
      int col = lane & 15, rb = (lane >> 4) << 3;
      #pragma unroll
      for (int j = 0; j < 8; ++j) {
        int row = rb + j;
        ATT[row * 16 + col] = (col <= row) ? acc[j] * EC[row] : 0.f; // ec*attn*tril
      }
    }
    if (tid >= 128 && tid < 256) {
      int d = tid - 128;
      float run = 0.f;
      #pragma unroll
      for (int i = 0; i < CHUNK; ++i) { run += G[i * DD + d]; CUM[i * DD + d] = run; }
    }
    __syncthreads();

    // ---- phase 5: zbar = qc@H - (ec*attn)@g + hb - ec*cumsum(g) ----
    {
      int col = 16 * w + (lane & 15);
      // (ec*attn) @ g : K=16 padded to 32
      v16h aAtt;
      {
        int row = lane & 15;
        #pragma unroll
        for (int i = 0; i < 16; ++i) aAtt[i] = (lane < 16) ? (_Float16)ATT[row * 16 + i] : (_Float16)0.f;
      }
      v16h bG = {};
      if (lane < 16) bG = *(const v16h*)(Gh + col * 16);

      int aRow = (lane & 15), aK = (lane >> 4) << 4, bK = (lane >> 4) << 4;
      v16h af[4], bf[4];
      #pragma unroll
      for (int t = 0; t < 4; ++t) {
        const _Float16* ap = QCh + aRow * DD + 32 * t + aK;
        af[t] = mkv16(*(const v8h*)ap, *(const v8h*)(ap + 8));
        bf[t] = *(const v16h*)(Hh + col * DD + 32 * t + bK);
      }
      v8f accM = (v8f){0.f,0.f,0.f,0.f,0.f,0.f,0.f,0.f};
      accM = wmma_f16(aAtt, bG, accM);
      v8f accQ = (v8f){0.f,0.f,0.f,0.f,0.f,0.f,0.f,0.f};
      #pragma unroll
      for (int t = 0; t < 4; ++t) accQ = wmma_f16(af[t], bf[t], accQ);

      int rb = (lane >> 4) << 3;
      #pragma unroll
      for (int j = 0; j < 8; ++j) {
        int row = rb + j;
        Zb[row * DD + col] = accQ[j] - accM[j] + HB[col] - EC[row] * CUM[row * DD + col];
      }
    }
    __syncthreads();

    // ---- phase 6: o = LN(zbar) + qc -> global f16 ----
    #pragma unroll
    for (int rr = 0; rr < 2; ++rr) {
      int r = 2 * w + rr;
      float s1 = 0.f, s2 = 0.f;
      for (int c = lane; c < DD; c += 32) { float z = Zb[r * DD + c]; s1 += z; s2 += z * z; }
      s1 = wred(s1); s2 = wred(s2);
      float mu = s1 * (1.f / DD);
      float var = s2 * (1.f / DD) - mu * mu;
      float rstd = rsqrtf(var + EPS_LN);
      size_t obase = (size_t)(b * TT + ch * CHUNK + r) * CC + h * DD;
      for (int c = lane; c < DD; c += 32) {
        float xh = (Zb[r * DD + c] - mu) * rstd;
        float o = GW[c] * xh + GB[c] + QC[r * DD + c];
        o_h[obase + c] = (_Float16)o;
      }
    }

    // ---- phase 7: H -= e_last * kc^T @ g ; hb -= e_last*sum(g) ----
    {
      float el = EC[CHUNK - 1];
      int m = 16 * w + (lane & 15); // this lane's A row (d index)
      v16h aK;
      {
        #pragma unroll
        for (int i = 0; i < 16; ++i) aK[i] = (lane < 16) ? KCh[i * DD + m] : (_Float16)0.f; // kc^T (already f16)
      }
      #pragma unroll
      for (int nt = 0; nt < 8; ++nt) {
        int n = nt * 16 + (lane & 15);
        v16h bG = {};
        if (lane < 16) bG = *(const v16h*)(Gh + n * 16);
        v8f acc = (v8f){0.f,0.f,0.f,0.f,0.f,0.f,0.f,0.f};
        acc = wmma_f16(aK, bG, acc);
        int mb = 16 * w + ((lane >> 4) << 3);
        #pragma unroll
        for (int j = 0; j < 8; ++j) {
          int kr = mb + j;
          float nv = Hs[kr * DD + n] - el * acc[j];
          Hs[kr * DD + n] = nv;
          Hh[n * DD + kr] = (_Float16)nv; // keep H^T (f16) in sync
        }
      }
      if (tid < 128) {
        float s = 0.f;
        #pragma unroll
        for (int i = 0; i < CHUNK; ++i) s += G[i * DD + tid];
        HB[tid] -= el * s;
      }
    }
    __syncthreads();
  }
}

// ---------------- launch ----------------
static inline size_t alignup(size_t v, size_t a) { return (v + a - 1) & ~(a - 1); }

extern "C" void kernel_launch(void* const* d_in, const int* in_sizes, int n_in,
                              void* d_out, int out_size, void* d_ws, size_t ws_size,
                              hipStream_t stream) {
  const float* x     = (const float*)d_in[0];
  const float* Wqkv  = (const float*)d_in[1];
  const float* bqkv  = (const float*)d_in[2];
  const float* Wproj = (const float*)d_in[3];
  const float* bproj = (const float*)d_in[4];
  const float* Wgate = (const float*)d_in[5];
  const float* ln_w  = (const float*)d_in[6];
  const float* ln_b  = (const float*)d_in[7];
  const float* h0    = (const float*)d_in[8];
  const float* hb0   = (const float*)d_in[9];
  float* out = (float*)d_out;

  const size_t MT = (size_t)BB * TT; // 8192
  char* p = (char*)d_ws;
  _Float16* x_h = (_Float16*)p;     p += alignup(MT * CC * 2, 256);
  _Float16* wqkv_h = (_Float16*)p;  p += alignup((size_t)3 * CC * CC * 2, 256);
  _Float16* wproj_h = (_Float16*)p; p += alignup((size_t)CC * CC * 2, 256);
  float* qkv = (float*)p;           p += alignup(MT * 3 * CC * 4, 256);
  float* eta = (float*)p;           p += alignup(MT * HH * 4, 256);
  _Float16* o_h = (_Float16*)p;     p += alignup(MT * CC * 2, 256);

  // f32 -> f16 staging
  cvt_f32_f16<<<4096, 256, 0, stream>>>(x, x_h, MT * CC);
  cvt_f32_f16<<<4096, 256, 0, stream>>>(Wqkv, wqkv_h, (size_t)3 * CC * CC);
  cvt_f32_f16<<<2048, 256, 0, stream>>>(Wproj, wproj_h, (size_t)CC * CC);

  // eta = ETA_BASE * sigmoid(x @ Wgate^T): one block per row
  eta_kernel<<<(int)MT, 256, 0, stream>>>(x, Wgate, eta);

  // qkv = x @ Wqkv^T + bqkv : (8192 x 6144)
  {
    dim3 grid((3 * CC) / 128, (int)(MT / 128));
    wmma_gemm<<<grid, 256, 0, stream>>>(x_h, wqkv_h, bqkv, qkv, (int)MT, 3 * CC, CC);
  }

  // sequential TTT scan, 1 workgroup per (b,h)
  {
    size_t smem = (size_t)(16384 + 6 * 2048 + 3 * 128 + 256 + 16) * sizeof(float)
                + (size_t)(16384 + 3 * 2048) * sizeof(_Float16);
    ttt_scan<<<BB * HH, 256, smem, stream>>>(qkv, eta, ln_w, ln_b, h0, hb0, o_h);
  }

  // out = o @ Wproj^T + bproj : (8192 x 2048)
  {
    dim3 grid(CC / 128, (int)(MT / 128));
    wmma_gemm<<<grid, 256, 0, stream>>>(o_h, wproj_h, bproj, out, (int)MT, CC, CC);
  }
}
